// LatentSpaceMemoryUnit_46712064311890
// MI455X (gfx1250) — matrix operand order repaired and checked by reading, not compiled, not run on hardware
//
#include <hip/hip_runtime.h>
#include <stdint.h>

#define N_ROWS  131072
#define FEA_DIM 256
#define MEM_DIM 512

typedef float   v8f   __attribute__((ext_vector_type(8)));
typedef __bf16  bf16_t;
typedef bf16_t  v16bf __attribute__((ext_vector_type(16)));
typedef bf16_t  v8bf  __attribute__((ext_vector_type(8)));

#define WMMA_BF16(a, b, c) \
  __builtin_amdgcn_wmma_f32_16x16x32_bf16(false, (a), false, (b), (short)0, (c), false, false)

// ---------- CDNA5 async global->LDS (ASYNCcnt-tracked, zero VGPR payload) ----------
__device__ __forceinline__ uint32_t lds_addr_of(const void* p) {
  return (uint32_t)(uintptr_t)p;   // addrspace(3) pointers are LDS byte offsets
}
// GVS form: uniform 64-bit base in SGPR pair, 32-bit per-thread offset in VGPR.
// No "memory" clobber: ordering is enforced by s_wait_asynccnt + barriers.
__device__ __forceinline__ void async_g2l_b128(uint32_t lds, const void* sbase, uint32_t voff) {
  asm volatile("global_load_async_to_lds_b128 %0, %1, %2"
               :: "v"(lds), "v"(voff), "s"(sbase));
}
__device__ __forceinline__ void wait_async0() {
  asm volatile("s_wait_asynccnt 0x0" ::: "memory");
}

// ---------- cross-lane reductions via ds_swizzle (wave32) ----------
__device__ __forceinline__ float half16_sum(float v) {
  int t;
  t = __builtin_amdgcn_ds_swizzle(__builtin_bit_cast(int, v), (1 << 10) | 0x1f);  v += __builtin_bit_cast(float, t);
  t = __builtin_amdgcn_ds_swizzle(__builtin_bit_cast(int, v), (2 << 10) | 0x1f);  v += __builtin_bit_cast(float, t);
  t = __builtin_amdgcn_ds_swizzle(__builtin_bit_cast(int, v), (4 << 10) | 0x1f);  v += __builtin_bit_cast(float, t);
  t = __builtin_amdgcn_ds_swizzle(__builtin_bit_cast(int, v), (8 << 10) | 0x1f);  v += __builtin_bit_cast(float, t);
  return v;  // sum over each 16-lane half
}
__device__ __forceinline__ float wave32_sum(float v) {
  v = half16_sum(v);
  int t = __builtin_amdgcn_ds_swizzle(__builtin_bit_cast(int, v), (16 << 10) | 0x1f);
  v += __builtin_bit_cast(float, t);
  return v;
}

// ---------- fragment loaders ----------
// B fragment (32x16 bf16): lane holds col = lane%16, K = e + 16*(lane/16); 16 contiguous bf16.
__device__ __forceinline__ v16bf load_bf16x16(const bf16_t* p) {
  const v8bf* q = (const v8bf*)p;
  v8bf lo = q[0], hi = q[1];
  return __builtin_shufflevector(lo, hi, 0, 1, 2, 3, 4, 5, 6, 7, 8, 9, 10, 11, 12, 13, 14, 15);
}
// A fragment from f32 row data: two contiguous 8-float runs -> 16 bf16
__device__ __forceinline__ v16bf cvt_a_frag(const float* run0, const float* run1) {
  const float4* p0 = (const float4*)run0;
  const float4* p1 = (const float4*)run1;
  float4 a0 = p0[0], a1 = p0[1], b0 = p1[0], b1 = p1[1];
  v16bf r;
  r[0]  = (bf16_t)a0.x; r[1]  = (bf16_t)a0.y; r[2]  = (bf16_t)a0.z; r[3]  = (bf16_t)a0.w;
  r[4]  = (bf16_t)a1.x; r[5]  = (bf16_t)a1.y; r[6]  = (bf16_t)a1.z; r[7]  = (bf16_t)a1.w;
  r[8]  = (bf16_t)b0.x; r[9]  = (bf16_t)b0.y; r[10] = (bf16_t)b0.z; r[11] = (bf16_t)b0.w;
  r[12] = (bf16_t)b1.x; r[13] = (bf16_t)b1.y; r[14] = (bf16_t)b1.z; r[15] = (bf16_t)b1.w;
  return r;
}

// ---------- chunk staging: columns [32k,32k+32) of W, as [rows][64B] in LDS ----------
__device__ __forceinline__ void stage_b1(uint32_t ldsbase, const bf16_t* Wbf, int ks, int tid) {
#pragma unroll
  for (int i = 0; i < 8; ++i) {               // 512 rows * 64B = 32KB / (256 thr * 16B)
    int idx = tid + 256 * i;
    int row = idx >> 2, q = idx & 3;
    async_g2l_b128(ldsbase + idx * 16, Wbf, (uint32_t)(row * 512 + ks * 64 + q * 16));
  }
}
__device__ __forceinline__ void stage_b2(uint32_t ldsbase, const bf16_t* WTbf, int ks, int tid) {
#pragma unroll
  for (int i = 0; i < 4; ++i) {               // 256 rows * 64B = 16KB
    int idx = tid + 256 * i;
    int row = idx >> 2, q = idx & 3;
    async_g2l_b128(ldsbase + idx * 16, WTbf, (uint32_t)(row * 1024 + ks * 64 + q * 16));
  }
}

// ---------- prep: inverse L2 norms of 256-wide rows (one wave per row) ----------
__global__ void rownorm_inv_kernel(const float* __restrict__ a, float* __restrict__ inv, int nrows) {
  int row  = blockIdx.x * (blockDim.x >> 5) + (threadIdx.x >> 5);
  int lane = threadIdx.x & 31;
  if (row >= nrows) return;
  float s = 0.f;
  const float* p = a + (size_t)row * FEA_DIM;
#pragma unroll
  for (int i = 0; i < 8; ++i) { float v = p[lane + 32 * i]; s += v * v; }
  s = wave32_sum(s);
  if (lane == 0)
    inv[row] = __builtin_amdgcn_rcpf(fmaxf(__builtin_amdgcn_sqrtf(s), 1e-8f));
}

// ---------- prep: W -> bf16 row-major [512][256] and transposed [256][512] ----------
__global__ void convert_w_kernel(const float* __restrict__ W, bf16_t* __restrict__ Wbf,
                                 bf16_t* __restrict__ WTbf) {
  int idx = blockIdx.x * 256 + threadIdx.x;     // 512*256 elements, grid=512
  int r = idx >> 8, c = idx & 255;
  bf16_t b = (bf16_t)W[idx];
  Wbf[idx] = b;
  WTbf[(size_t)c * MEM_DIM + r] = b;
}

// ---------- fused main kernel: 8 waves/block, one 16-row tile per wave ----------
__global__ void __launch_bounds__(256)
fused_kernel(const float* __restrict__ x, const bf16_t* __restrict__ Wbf,
             const bf16_t* __restrict__ WTbf, const float* __restrict__ xinv,
             const float* __restrict__ winv, float* __restrict__ out,
             float* __restrict__ att_out, float* __restrict__ pre_out) {
  extern __shared__ char smem[];
  bf16_t* smB    = (bf16_t*)smem;               // 2 x 32KB B ping-pong (GEMM2: 2 x 16KB)
  float*  smXall = (float*)(smem + 65536);      // 8 x 16KB per-wave x tiles / att bf16

  const int tid  = threadIdx.x;
  const int w    = tid >> 5;
  const int lane = tid & 31;
  const int c    = lane & 15;   // tile column / A row
  const int h    = lane >> 4;   // lane half
  const int rowBase = blockIdx.x * 128 + w * 16;
  float*  smX = smXall + w * (16 * FEA_DIM);
  bf16_t* smA = (bf16_t*)smX;                   // att bf16 alias (same 16KB)
  const uint32_t smB_lds = lds_addr_of(smB);

  // ---- issue async stages: per-wave x tile (16KB) + block-wide B chunk 0 ----
  {
    const char* xblock = (const char*)(x + (size_t)blockIdx.x * 128 * FEA_DIM); // uniform
    uint32_t off = (uint32_t)(w * 16384 + lane * 16);
    uint32_t dst = lds_addr_of(smXall) + off;
#pragma unroll
    for (int i = 0; i < 32; ++i) async_g2l_b128(dst + i * 512, xblock, off + i * 512);
  }
  stage_b1(smB_lds, Wbf, 0, tid);

  float rxv[8];
#pragma unroll
  for (int r = 0; r < 8; ++r) rxv[r] = xinv[rowBase + r + 8 * h];

  // ---- GEMM1: logits[16][512] = x_tile @ W^T  (bf16 WMMA, f32 acc) ----
  v8f acc[32];
#pragma unroll
  for (int n = 0; n < 32; ++n)
    acc[n] = (v8f){0.f, 0.f, 0.f, 0.f, 0.f, 0.f, 0.f, 0.f};

#pragma unroll 1
  for (int ks = 0; ks < 8; ++ks) {                 // K = 256 in steps of 32
    wait_async0();
    __syncthreads();                               // chunk ks (and x tile) resident
    if (ks < 7) stage_b1(smB_lds + ((ks + 1) & 1) * 32768, Wbf, ks + 1, tid);
    const bf16_t* bbuf = smB + (ks & 1) * 16384;
    const float* rowp = smX + c * FEA_DIM;
    v16bf a = cvt_a_frag(rowp + 32 * ks + 16 * h, rowp + 32 * ks + 16 * h + 8);
#pragma unroll
    for (int g = 0; g < 8; ++g) {
      const bf16_t* bp = bbuf + (64 * g + c) * 32 + 16 * h;   // n = 4g+j -> +j*512
      v16bf b0 = load_bf16x16(bp + 0 * 512);
      v16bf b1 = load_bf16x16(bp + 1 * 512);
      v16bf b2 = load_bf16x16(bp + 2 * 512);
      v16bf b3 = load_bf16x16(bp + 3 * 512);
      acc[4 * g + 0] = WMMA_BF16(a, b0, acc[4 * g + 0]);
      acc[4 * g + 1] = WMMA_BF16(a, b1, acc[4 * g + 1]);
      acc[4 * g + 2] = WMMA_BF16(a, b2, acc[4 * g + 2]);
      acc[4 * g + 3] = WMMA_BF16(a, b3, acc[4 * g + 3]);
    }
    // IGLP recipe: 68 DS reads (4 A + 64 B) / 32 WMMAs, loads kept ~2 frags ahead.
    __builtin_amdgcn_sched_group_barrier(0x100, 8, 0);   // DS_READ x8
#pragma unroll
    for (int it = 0; it < 15; ++it) {
      __builtin_amdgcn_sched_group_barrier(0x008, 2, 0); // WMMA x2
      __builtin_amdgcn_sched_group_barrier(0x100, 4, 0); // DS_READ x4
    }
    __builtin_amdgcn_sched_group_barrier(0x008, 2, 0);   // WMMA x2
  }

  // ---- scale to cosine sim, store pre_softmax, exp (no max pass: |cos|<=1) ----
  float psum[8];
#pragma unroll
  for (int r = 0; r < 8; ++r) psum[r] = 0.f;
#pragma unroll
  for (int n = 0; n < 32; ++n) {
    int col = 16 * n + c;
    float wv = winv[col];
#pragma unroll
    for (int r = 0; r < 8; ++r) {
      float logit = acc[n][r] * rxv[r] * wv;
      pre_out[(size_t)(rowBase + r + 8 * h) * MEM_DIM + col] = logit;
      float e = __builtin_amdgcn_exp2f(logit * 1.4426950408889634f);
      acc[n][r] = e;
      psum[r] += e;
    }
  }
  float sinv[8];
#pragma unroll
  for (int r = 0; r < 8; ++r) sinv[r] = __builtin_amdgcn_rcpf(half16_sum(psum[r]));

  // ---- hard_shrink_relu + L1 accumulation ----
  float l1[8];
#pragma unroll
  for (int r = 0; r < 8; ++r) l1[r] = 0.f;
#pragma unroll
  for (int n = 0; n < 32; ++n) {
#pragma unroll
    for (int r = 0; r < 8; ++r) {
      float att  = acc[n][r] * sinv[r];
      float sh   = att - 0.0025f;
      float keep = fmaxf(sh, 0.f) * att * __builtin_amdgcn_rcpf(fabsf(sh) + 1e-12f);
      acc[n][r] = keep;
      l1[r] += keep;
    }
  }
  float l1inv[8];
#pragma unroll
  for (int r = 0; r < 8; ++r)
    l1inv[r] = __builtin_amdgcn_rcpf(fmaxf(half16_sum(l1[r]), 1e-12f));

  // ---- finalize att: f32 to global, bf16 to per-wave LDS in WMMA-A row-major ----
  asm volatile("" ::: "memory");
#pragma unroll
  for (int n = 0; n < 32; ++n) {
    int col = 16 * n + c;
#pragma unroll
    for (int r = 0; r < 8; ++r) {
      float a = acc[n][r] * l1inv[r];
      att_out[(size_t)(rowBase + r + 8 * h) * MEM_DIM + col] = a;
      smA[(r + 8 * h) * MEM_DIM + col] = (bf16_t)a;
    }
  }
  asm volatile("" ::: "memory");

  // ---- GEMM2: out[16][256] = att[16][512] @ W[512][256] ----
  __syncthreads();                 // everyone done reading GEMM1 B chunks
  stage_b2(smB_lds, WTbf, 0, tid);

  v8f acc2[16];
#pragma unroll
  for (int n = 0; n < 16; ++n)
    acc2[n] = (v8f){0.f, 0.f, 0.f, 0.f, 0.f, 0.f, 0.f, 0.f};

#pragma unroll 1
  for (int ks = 0; ks < 16; ++ks) {                // K = 512 in steps of 32
    wait_async0();
    __syncthreads();
    if (ks < 15) stage_b2(smB_lds + ((ks + 1) & 1) * 16384, WTbf, ks + 1, tid);
    const bf16_t* bbuf = smB + (ks & 1) * 8192;
    const v8bf* q0 = (const v8bf*)(smA + c * MEM_DIM + 32 * ks + 16 * h);
    const v8bf* q1 = (const v8bf*)(smA + c * MEM_DIM + 32 * ks + 16 * h + 8);
    v16bf a2 = __builtin_shufflevector(q0[0], q1[0], 0, 1, 2, 3, 4, 5, 6, 7,
                                       8, 9, 10, 11, 12, 13, 14, 15);
#pragma unroll
    for (int g = 0; g < 4; ++g) {
      const bf16_t* bp = bbuf + (64 * g + c) * 32 + 16 * h;
      v16bf b0 = load_bf16x16(bp + 0 * 512);
      v16bf b1 = load_bf16x16(bp + 1 * 512);
      v16bf b2 = load_bf16x16(bp + 2 * 512);
      v16bf b3 = load_bf16x16(bp + 3 * 512);
      acc2[4 * g + 0] = WMMA_BF16(a2, b0, acc2[4 * g + 0]);
      acc2[4 * g + 1] = WMMA_BF16(a2, b1, acc2[4 * g + 1]);
      acc2[4 * g + 2] = WMMA_BF16(a2, b2, acc2[4 * g + 2]);
      acc2[4 * g + 3] = WMMA_BF16(a2, b3, acc2[4 * g + 3]);
    }
    // IGLP recipe: 34 DS reads (2 A + 32 B) / 16 WMMAs.
    __builtin_amdgcn_sched_group_barrier(0x100, 6, 0);   // DS_READ x6
#pragma unroll
    for (int it = 0; it < 7; ++it) {
      __builtin_amdgcn_sched_group_barrier(0x008, 2, 0); // WMMA x2
      __builtin_amdgcn_sched_group_barrier(0x100, 4, 0); // DS_READ x4
    }
    __builtin_amdgcn_sched_group_barrier(0x008, 2, 0);   // WMMA x2
  }
#pragma unroll
  for (int n2 = 0; n2 < 16; ++n2) {
    int col = 16 * n2 + c;
#pragma unroll
    for (int r = 0; r < 8; ++r)
      out[(size_t)(rowBase + r + 8 * h) * FEA_DIM + col] = acc2[n2][r];
  }
}

extern "C" void kernel_launch(void* const* d_in, const int* in_sizes, int n_in,
                              void* d_out, int out_size, void* d_ws, size_t ws_size,
                              hipStream_t stream) {
  (void)in_sizes; (void)n_in; (void)out_size; (void)ws_size;
  const float* x = (const float*)d_in[0];   // [N, 256]
  const float* W = (const float*)d_in[1];   // [512, 256]

  float* out     = (float*)d_out;                       // [N,256]
  float* att_out = out + (size_t)N_ROWS * FEA_DIM;      // [N,512]
  float* pre_out = att_out + (size_t)N_ROWS * MEM_DIM;  // [N,512]

  // workspace layout (~1.05 MB)
  char* ws = (char*)d_ws;
  float*  xinv = (float*)ws;                                  // N floats
  float*  winv = (float*)(ws + (size_t)N_ROWS * 4);           // 512 floats
  bf16_t* Wbf  = (bf16_t*)(ws + (size_t)N_ROWS * 4 + 2048);   // [512][256] bf16
  bf16_t* WTbf = Wbf + (size_t)MEM_DIM * FEA_DIM;             // [256][512] bf16

  convert_w_kernel<<<512, 256, 0, stream>>>(W, Wbf, WTbf);
  rownorm_inv_kernel<<<N_ROWS / 8, 256, 0, stream>>>(x, xinv, N_ROWS);
  rownorm_inv_kernel<<<MEM_DIM / 8, 256, 0, stream>>>(W, winv, MEM_DIM);

  const size_t lds_bytes = 65536 + 131072;   // 64KB B ping-pong + 8x16KB wave tiles
  fused_kernel<<<N_ROWS / 128, 256, lds_bytes, stream>>>(x, Wbf, WTbf, xinv, winv,
                                                         out, att_out, pre_out);
}
